// TSNE_NN_18150531793122
// MI455X (gfx1250) — matrix operand order repaired and checked by reading, not compiled, not run on hardware
//
#include <hip/hip_runtime.h>
#include <hip/hip_bf16.h>
#include <math.h>

#define B_PTS 8192
#define N_PTS 10000
#define D_IN  64
#define H_DIM 5000
#define EPSF  1e-36f

typedef __attribute__((ext_vector_type(16))) __bf16 v16bf;
typedef __attribute__((ext_vector_type(8)))  __bf16 v8bf;
typedef __attribute__((ext_vector_type(8)))  float  v8f;

static __device__ __forceinline__ v16bf cat8(v8bf lo, v8bf hi) {
  return __builtin_shufflevector(lo, hi, 0,1,2,3,4,5,6,7,8,9,10,11,12,13,14,15);
}

// Low 32 bits of a generic LDS pointer == wave-relative LDS byte offset.
static __device__ __forceinline__ unsigned lds_off(const void* p) {
  return (unsigned)(uintptr_t)p;
}

// ---------------- Kernel 0: zero accumulators + convert W1 -> bf16 transposed [H][D]
__global__ void __launch_bounds__(256)
k0_init(const float* __restrict__ W1, __bf16* __restrict__ W1T, float* __restrict__ acc) {
  int t = blockIdx.x * blockDim.x + threadIdx.x;
  if (blockIdx.x == 0 && threadIdx.x < 4) acc[threadIdx.x] = 0.0f;
  if (t < D_IN * H_DIM) {
    int n = t / D_IN;
    int k = t - n * D_IN;
    W1T[n * D_IN + k] = (__bf16)W1[(size_t)k * H_DIM + n];
  }
}

// ---------------- Kernel 1: fused gather + sigmoid(x@W1+b1) @ W2 + b2 -> y[B,2], ysq[B]
// Block = 16 batch rows, 4 waves. W1T streamed through double-buffered LDS tiles via
// global_load_async_to_lds_b128 (ASYNCcnt), WMMA consumes from LDS; group loop is
// block-uniform so EXEC is all-ones at every WMMA.
__global__ void __launch_bounds__(128)
k1_mlp(const float* __restrict__ data, const __bf16* __restrict__ W1T,
       const float* __restrict__ b1, const float* __restrict__ W2,
       const float* __restrict__ b2, const int* __restrict__ sidx,
       float* __restrict__ y, float* __restrict__ ysq)
{
  __shared__ __align__(16) float  xsf[16][64];      // gathered x rows (raw f32)
  __shared__ __align__(16) __bf16 w1s[2][64][64];   // double-buffered W1T group tiles (16KB)
  __shared__ float yacc[16][2];

  const int tid = threadIdx.x;
  const int i0  = blockIdx.x * 16;
  if (tid < 32) yacc[tid >> 1][tid & 1] = 0.0f;

  const unsigned long long dbase = (unsigned long long)(uintptr_t)data;
  const unsigned long long wbase = (unsigned long long)(uintptr_t)W1T;

  // Stage one 64-column group of W1T into w1s[buf] : 4 async b128 per thread.
  auto stage = [&](int g, int buf) {
    #pragma unroll
    for (int i = 0; i < 4; ++i) {
      int idx = tid + i * 128;            // 0..511
      int r   = idx >> 3;                 // local column 0..63
      int sg  = idx & 7;                  // 16B segment within 128B row
      int col = g * 64 + r;
      if (col >= H_DIM) col = 0;          // safe source; masked later via w2=0
      unsigned goff = (unsigned)col * (D_IN * 2) + (unsigned)sg * 16;
      unsigned loff = lds_off(&w1s[buf][r][sg * 8]);
      asm volatile("global_load_async_to_lds_b128 %0, %1, %2 offset:0"
                   :: "v"(loff), "v"(goff), "s"(wbase) : "memory");
    }
  };

  // ---- async gather of x tile: 16 rows x 256B = 2 async b128 per thread
  #pragma unroll
  for (int i = 0; i < 2; ++i) {
    int idx = tid + i * 128;              // 0..255
    int r   = idx >> 4;                   // row 0..15
    int sg  = idx & 15;                   // 16B segment within 256B row
    int si  = sidx[i0 + r];
    unsigned goff = (unsigned)si * (D_IN * 4) + (unsigned)sg * 16;
    unsigned loff = lds_off(&xsf[r][sg * 4]);
    asm volatile("global_load_async_to_lds_b128 %0, %1, %2 offset:0"
                 :: "v"(loff), "v"(goff), "s"(dbase) : "memory");
  }

  const int NG = (H_DIM + 63) / 64;       // 79 groups (64 columns each)
  stage(0, 0);                            // prefetch first W1T group

  // x loads (issued first) complete once <=4 (the group-0 stage) remain in flight
  asm volatile("s_wait_asynccnt 0x4" ::: "memory");
  __syncthreads();

  const int lane = tid & 31;
  const int wave = tid >> 5;
  const int half = lane >> 4;             // 0 or 1
  const int mrow = lane & 15;             // M row (A) / N col (B,C)
  const int k0   = half * 8;              // A-frag K base
  const int kg   = half * 16;             // unused (B frag is built with k0)

  // Build A fragments (16x32 bf16 each, lane layout per ISA 7.12.2) from LDS f32.
  v16bf a0, a1;
  #pragma unroll
  for (int e = 0; e < 8; ++e) {
    a0[e]     = (__bf16)xsf[mrow][k0 + e];
    a0[e + 8] = (__bf16)xsf[mrow][k0 + 16 + e];
    a1[e]     = (__bf16)xsf[mrow][k0 + 32 + e];
    a1[e + 8] = (__bf16)xsf[mrow][k0 + 48 + e];
  }

  float acc0[8], acc1[8];
  #pragma unroll
  for (int r = 0; r < 8; ++r) { acc0[r] = 0.0f; acc1[r] = 0.0f; }

  for (int g = 0; g < NG; ++g) {
    const int buf = g & 1;
    if (g + 1 < NG) {
      stage(g + 1, buf ^ 1);
      asm volatile("s_wait_asynccnt 0x4" ::: "memory");  // group g landed
    } else {
      asm volatile("s_wait_asynccnt 0x0" ::: "memory");
    }
    __syncthreads();

    const int coln = g * 64 + wave * 16 + mrow;          // this lane's column
    const v8bf* wp = reinterpret_cast<const v8bf*>(&w1s[buf][wave * 16 + mrow][0]);
    v16bf bA = cat8(wp[(k0)      >> 3], wp[(16 + k0) >> 3]);        // K 0..31
    v16bf bB = cat8(wp[(32 + k0) >> 3], wp[(48 + k0) >> 3]);   // K 32..63

    float w20 = 0.0f, w21 = 0.0f, bb1 = 0.0f;
    if (coln < H_DIM) {
      w20 = W2[coln * 2 + 0];
      w21 = W2[coln * 2 + 1];
      bb1 = b1[coln];
    }

    v8f cf = {};
    cf = __builtin_amdgcn_wmma_f32_16x16x32_bf16(false, a0, false, bA, (short)0, cf, false, false);
    cf = __builtin_amdgcn_wmma_f32_16x16x32_bf16(false, a1, false, bB, (short)0, cf, false, false);

    #pragma unroll
    for (int r = 0; r < 8; ++r) {
      float h = __builtin_amdgcn_rcpf(1.0f + __expf(-(cf[r] + bb1)));  // sigmoid
      acc0[r] += h * w20;
      acc1[r] += h * w21;
    }
    __syncthreads();   // all waves done reading w1s[buf] before it is restaged
  }

  // Reduce over the 16 N-lanes within each half-wave
  #pragma unroll
  for (int m = 1; m < 16; m <<= 1) {
    #pragma unroll
    for (int r = 0; r < 8; ++r) {
      acc0[r] += __shfl_xor(acc0[r], m, 32);
      acc1[r] += __shfl_xor(acc1[r], m, 32);
    }
  }
  if (mrow == 0) {
    #pragma unroll
    for (int r = 0; r < 8; ++r) {
      atomicAdd(&yacc[r + half * 8][0], acc0[r]);
      atomicAdd(&yacc[r + half * 8][1], acc1[r]);
    }
  }
  __syncthreads();

  if (tid < 16) {
    float y0 = yacc[tid][0] + b2[0];
    float y1 = yacc[tid][1] + b2[1];
    y[(i0 + tid) * 2 + 0] = y0;
    y[(i0 + tid) * 2 + 1] = y1;
    ysq[i0 + tid] = y0 * y0 + y1 * y1;
  }
}

// ---------------- Kernel 2: sum_m = sum_{i!=j} 1/(1+dis_ij)
__global__ void __launch_bounds__(256)
k2_summ(const float* __restrict__ y, const float* __restrict__ ysq, float* acc) {
  __shared__ float sy0[256], sy1[256], ss[256], red[256];
  const int tid = threadIdx.x;
  const int i = blockIdx.x * 256 + tid;
  const float yi0 = y[i * 2 + 0], yi1 = y[i * 2 + 1], yiq = ysq[i];
  float s = 0.0f;
  for (int jt = 0; jt < B_PTS; jt += 256) {
    __syncthreads();
    sy0[tid] = y[(jt + tid) * 2 + 0];
    sy1[tid] = y[(jt + tid) * 2 + 1];
    ss[tid]  = ysq[jt + tid];
    __syncthreads();
    #pragma unroll 4
    for (int jj = 0; jj < 256; ++jj) {
      float d = yiq + ss[jj] - 2.0f * (yi0 * sy0[jj] + yi1 * sy1[jj]);
      d = fmaxf(d, EPSF);
      float q = __builtin_amdgcn_rcpf(1.0f + d);
      s += ((jt + jj) == i) ? 0.0f : q;   // excludes trace
    }
  }
  red[tid] = s;
  __syncthreads();
  for (int st = 128; st > 0; st >>= 1) {
    if (tid < st) red[tid] += red[tid + st];
    __syncthreads();
  }
  if (tid == 0) atomicAdd(&acc[0], red[0]);
}

// ---------------- Kernel 3: KL + PL partials (gathered pij/near reads dominate: ~536MB)
__global__ void __launch_bounds__(256)
k3_loss(const float* __restrict__ y, const float* __restrict__ ysq,
        const float* __restrict__ pij, const float* __restrict__ nearin,
        const int* __restrict__ sidx, const float* acc_in, float* acc_out)
{
  __shared__ float jy0[1024], jy1[1024], jsq[1024];
  __shared__ int   jsi[1024];
  __shared__ float r0[256], r1[256], r2[256];

  const int tid   = threadIdx.x;
  const int irow  = blockIdx.x * 16 + (tid >> 4);
  const int jlane = tid & 15;
  const int si = sidx[irow];
  const float* prow = pij    + (size_t)si * N_PTS;
  const float* nrow = nearin + (size_t)si * N_PTS;
  const float yi0 = y[irow * 2 + 0], yi1 = y[irow * 2 + 1], yiq = ysq[irow];
  const float inv_summ = 1.0f / acc_in[0];

  float kl = 0.0f, nrm2 = 0.0f, cnt = 0.0f;
  for (int jt = 0; jt < B_PTS; jt += 1024) {
    __syncthreads();
    for (int u = tid; u < 1024; u += 256) {
      int j = jt + u;
      jy0[u] = y[j * 2 + 0];
      jy1[u] = y[j * 2 + 1];
      jsq[u] = ysq[j];
      jsi[u] = sidx[j];
    }
    __syncthreads();
    for (int jj = jlane; jj < 1024; jj += 16) {
      float d = yiq + jsq[jj] - 2.0f * (yi0 * jy0[jj] + yi1 * jy1[jj]);
      d = fmaxf(d, EPSF);
      int sj = jsi[jj];
      float p  = prow[sj];
      float nr = nrow[sj];
      float q   = __builtin_amdgcn_rcpf(1.0f + d);
      float qij = fmaxf(q * inv_summ, EPSF);
      kl += p * (__logf(p) - __logf(qij));
      float e = fminf(d, 10.0f);
      bool far = (nr == 0.0f);
      nrm2 += far ? e * e : 0.0f;
      cnt  += far ? 1.0f : 0.0f;
    }
  }
  r0[tid] = kl; r1[tid] = nrm2; r2[tid] = cnt;
  __syncthreads();
  for (int st = 128; st > 0; st >>= 1) {
    if (tid < st) { r0[tid] += r0[tid + st]; r1[tid] += r1[tid + st]; r2[tid] += r2[tid + st]; }
    __syncthreads();
  }
  if (tid == 0) {
    atomicAdd(&acc_out[1], r0[0]);
    atomicAdd(&acc_out[2], r1[0]);
    atomicAdd(&acc_out[3], r2[0]);
  }
}

// ---------------- Kernel 4: finalize (pl, kl)
__global__ void k4_final(const float* __restrict__ acc, float* __restrict__ out) {
  if (threadIdx.x == 0 && blockIdx.x == 0) {
    out[0] = -500.0f * sqrtf(acc[2]) / acc[3];   // RATE_PL = 1
    out[1] = acc[1];                             // RATE_KL = 1
  }
}

extern "C" void kernel_launch(void* const* d_in, const int* in_sizes, int n_in,
                              void* d_out, int out_size, void* d_ws, size_t ws_size,
                              hipStream_t stream) {
  const float* data   = (const float*)d_in[0];
  const float* W1     = (const float*)d_in[1];
  const float* b1     = (const float*)d_in[2];
  const float* W2     = (const float*)d_in[3];
  const float* b2     = (const float*)d_in[4];
  const float* pij    = (const float*)d_in[5];
  const float* nearin = (const float*)d_in[6];
  const int*   sidx   = (const int*)d_in[7];
  float* out = (float*)d_out;

  char* ws = (char*)d_ws;
  float*  acc = (float*)(ws);                           // 4 floats
  float*  y   = (float*)(ws + 256);                     // 8192*2 f32 = 64KB
  float*  ysq = (float*)(ws + 256 + 65536);             // 8192 f32  = 32KB
  __bf16* W1T = (__bf16*)(ws + 256 + 65536 + 32768);    // 5000*64 bf16 = 640KB

  k0_init <<<(D_IN * H_DIM + 255) / 256, 256, 0, stream>>>(W1, W1T, acc);
  k1_mlp  <<<B_PTS / 16, 128, 0, stream>>>(data, W1T, b1, W2, b2, sidx, y, ysq);
  k2_summ <<<B_PTS / 256, 256, 0, stream>>>(y, ysq, acc);
  k3_loss <<<B_PTS / 16, 256, 0, stream>>>(y, ysq, pij, nearin, sidx, acc, acc);
  k4_final<<<1, 32, 0, stream>>>(acc, out);
}
